// TransformerEncoderLayerWithMem_14929306321519
// MI455X (gfx1250) — matrix operand order repaired and checked by reading, not compiled, not run on hardware
//
#include <hip/hip_runtime.h>
#include <hip/hip_bf16.h>
#include <math.h>

#define E_DIM 1024
#define H_NUM 16
#define HD    64
#define T_SEQ 2048
#define B_SZ  2
#define M_MEM 32
#define S_LEN (T_SEQ + M_MEM)   // 2080
#define F_DIM 4096
#define R_ROWS (T_SEQ * B_SZ)   // 4096

typedef __bf16 v16bf __attribute__((ext_vector_type(16)));
typedef __bf16 v8bf  __attribute__((ext_vector_type(8)));
typedef float  v8f   __attribute__((ext_vector_type(8)));

__device__ __forceinline__ __bf16 f2bf(float f) {
  unsigned int u = __float_as_uint(f);
  u += 0x7FFFu + ((u >> 16) & 1u);            // round-to-nearest-even
  unsigned short s = (unsigned short)(u >> 16);
  return __builtin_bit_cast(__bf16, s);
}

// ---- WMMA fragment loaders (wave32, contiguous k order in A and B) -------------
// A 16x32 bf16: lane<16 holds row m=lane, K 0..15; lane>=16 row
// m=lane-16, K 16..31 (contiguous).  Two 16B loads per lane.
__device__ __forceinline__ v16bf load_a_frag(const __bf16* base, int ld,
                                             int row0, int k0, int lane) {
  const int half = lane >> 4;
  const __bf16* p = base + (size_t)(row0 + (lane & 15)) * ld + k0 + half * 16;
  v8bf lo = *(const v8bf*)(p);
  v8bf hi = *(const v8bf*)(p + 8);
  v16bf f;
#pragma unroll
  for (int i = 0; i < 8; ++i) { f[i] = lo[i]; f[i + 8] = hi[i]; }
  return f;
}
// B 32x16 bf16: lane holds column n=lane&15; lanes<16 K=0..15, lanes>=16
// K=16..31 (contiguous).  Source stored row-major [N,K] (i.e. W for X@W^T).
__device__ __forceinline__ v16bf load_b_frag(const __bf16* base, int ld,
                                             int n0, int k0, int lane) {
  const __bf16* p = base + (size_t)(n0 + (lane & 15)) * ld + k0 + (lane >> 4) * 16;
  v8bf lo = *(const v8bf*)(p);
  v8bf hi = *(const v8bf*)(p + 8);
  v16bf f;
#pragma unroll
  for (int i = 0; i < 8; ++i) { f[i] = lo[i]; f[i + 8] = hi[i]; }
  return f;
}

__device__ __forceinline__ v8f wmma_bf16(v16bf a, v16bf b, v8f c) {
  return __builtin_amdgcn_wmma_f32_16x16x32_bf16(false, a, false, b,
                                                 (short)0, c, false, false);
}

// C = A[rows,K] @ W[cols,K]^T : one wave computes a 32x64 tile (8 C-frags).
// Two A-fragments share each B-fragment -> 12 b128 loads per 8 WMMAs.
__device__ __forceinline__ void gemm_tile_32x64(const __bf16* __restrict__ A, int lda, int row0,
                                                const __bf16* __restrict__ W, int ldw, int col0,
                                                int kdim, int lane, v8f acc[8]) {
#pragma unroll 2
  for (int k = 0; k < kdim; k += 32) {
    v16bf a0 = load_a_frag(A, lda, row0, k, lane);
    v16bf a1 = load_a_frag(A, lda, row0 + 16, k, lane);
#pragma unroll
    for (int t = 0; t < 4; ++t) {
      v16bf b = load_b_frag(W, ldw, col0 + 16 * t, k, lane);
      acc[t]     = wmma_bf16(a0, b, acc[t]);
      acc[t + 4] = wmma_bf16(a1, b, acc[t + 4]);
    }
  }
}

__device__ __forceinline__ float rowmax16(float v) {
  v = fmaxf(v, __shfl_xor(v, 1));
  v = fmaxf(v, __shfl_xor(v, 2));
  v = fmaxf(v, __shfl_xor(v, 4));
  v = fmaxf(v, __shfl_xor(v, 8));
  return v;
}
__device__ __forceinline__ float rowsum16(float v) {
  v += __shfl_xor(v, 1); v += __shfl_xor(v, 2);
  v += __shfl_xor(v, 4); v += __shfl_xor(v, 8);
  return v;
}

// ---- elementwise fp32 -> bf16 ---------------------------------------------
__global__ void f2bf_kernel(const float* __restrict__ in, __bf16* __restrict__ out, int n) {
  int i = blockIdx.x * blockDim.x + threadIdx.x;
  if (i < n) out[i] = f2bf(in[i]);
}

// ---- QKV GEMM + scatter to head layouts -----------------------------------
// Q: [B,H,T,64] (scaled by d^-1/2), K: [B,H,S,64], V^T: [B,H,64,S]
__global__ void __launch_bounds__(256)
qkv_kernel(const __bf16* __restrict__ A, const __bf16* __restrict__ W,
           const float* __restrict__ bias, __bf16* __restrict__ Qb,
           __bf16* __restrict__ Kb, __bf16* __restrict__ VTb) {
  const int lane = threadIdx.x & 31;
  const int wid  = blockIdx.x * 8 + (threadIdx.x >> 5);
  const int row0 = (wid / 48) * 32;
  const int col0 = (wid % 48) * 64;
  v8f acc[8] = {};
  gemm_tile_32x64(A, E_DIM, row0, W, E_DIM, col0, E_DIM, lane, acc);
  const int nrow = lane & 15, madd = (lane >> 4) * 8;
#pragma unroll
  for (int t = 0; t < 4; ++t) {
    const int c = col0 + 16 * t + nrow;
    const float bc = bias[c];
#pragma unroll
    for (int g = 0; g < 2; ++g) {
#pragma unroll
      for (int v = 0; v < 8; ++v) {
        const int r = row0 + 16 * g + v + madd;
        const int tq = r >> 1, bb = r & 1;
        float val = acc[g * 4 + t][v];
        val += bc;
        if (c < E_DIM) {
          const int h = c >> 6, dd = c & 63;
          Qb[((size_t)(bb * H_NUM + h) * T_SEQ + tq) * HD + dd] = f2bf(val * 0.125f);
        } else if (c < 2 * E_DIM) {
          const int c2 = c - E_DIM, h = c2 >> 6, dd = c2 & 63;
          Kb[((size_t)(bb * H_NUM + h) * S_LEN + tq) * HD + dd] = f2bf(val);
        } else {
          const int c2 = c - 2 * E_DIM, h = c2 >> 6, dd = c2 & 63;
          VTb[((size_t)(bb * H_NUM + h) * HD + dd) * S_LEN + tq] = f2bf(val);
        }
      }
    }
  }
}

// ---- append scaled memory tokens to K / V^T -------------------------------
__global__ void memkv_kernel(const float* __restrict__ mem_k, const float* __restrict__ mem_v,
                             __bf16* __restrict__ Kb, __bf16* __restrict__ VTb) {
  int i = blockIdx.x * blockDim.x + threadIdx.x;   // over M*E = 32768
  if (i >= M_MEM * E_DIM) return;
  int m = i >> 10, c = i & 1023, h = c >> 6, dd = c & 63;
  __bf16 kv = f2bf(8.0f * mem_k[i]);               // d^0.5 = 8
  __bf16 vv = f2bf(5.656854249f * mem_v[i]);       // M^0.5
#pragma unroll
  for (int bb = 0; bb < B_SZ; ++bb) {
    Kb[((size_t)(bb * H_NUM + h) * S_LEN + T_SEQ + m) * HD + dd] = kv;
    VTb[((size_t)(bb * H_NUM + h) * HD + dd) * S_LEN + T_SEQ + m] = vv;
  }
}

// ---- flash attention: one wave per 16-row Q tile --------------------------
__global__ void __launch_bounds__(128)
attn_kernel(const __bf16* __restrict__ Qb, const __bf16* __restrict__ Kb,
            const __bf16* __restrict__ VTb, __bf16* __restrict__ ctx) {
  const int lane = threadIdx.x & 31;
  const int wave = threadIdx.x >> 5;
  const int tb = blockIdx.x & 31;     // 32 blocks of 64 q-rows
  const int bh = blockIdx.x >> 5;     // b*16+h
  const int b = bh >> 4, h = bh & 15;
  const int q0 = tb * 64 + wave * 16;

  const __bf16* Qbh = Qb + (size_t)bh * T_SEQ * HD;
  const __bf16* Kbh = Kb + (size_t)bh * S_LEN * HD;
  const __bf16* Vbh = VTb + (size_t)bh * HD * S_LEN;

  __shared__ __align__(16) __bf16 plds[4][16 * 32];
  __bf16* myp = &plds[wave][0];

  v16bf qa0 = load_a_frag(Qbh, HD, q0, 0, lane);
  v16bf qa1 = load_a_frag(Qbh, HD, q0, 32, lane);

  v8f acc0 = {}, acc1 = {}, acc2 = {}, acc3 = {};
  float run_max[8], run_sum[8];
#pragma unroll
  for (int v = 0; v < 8; ++v) { run_max[v] = -3.0e38f; run_sum[v] = 0.0f; }

  const int nrow = lane & 15;
  const int madd = (lane >> 4) * 8;

  for (int s0 = 0; s0 < S_LEN; s0 += 32) {
    v8f sc0 = {}, sc1 = {};
    {
      v16bf kb = load_b_frag(Kbh, HD, s0, 0, lane);
      sc0 = wmma_bf16(qa0, kb, sc0);
      kb = load_b_frag(Kbh, HD, s0, 32, lane);
      sc0 = wmma_bf16(qa1, kb, sc0);
      kb = load_b_frag(Kbh, HD, s0 + 16, 0, lane);
      sc1 = wmma_bf16(qa0, kb, sc1);
      kb = load_b_frag(Kbh, HD, s0 + 16, 32, lane);
      sc1 = wmma_bf16(qa1, kb, sc1);
    }
#pragma unroll
    for (int v = 0; v < 8; ++v) {        // online softmax, row stats per lane-group
      float a = sc0[v], c = sc1[v];
      float tmax = rowmax16(fmaxf(a, c));
      float nm = fmaxf(run_max[v], tmax);
      float scale = __expf(run_max[v] - nm);
      run_max[v] = nm;
      float e0 = __expf(a - nm);
      float e1 = __expf(c - nm);
      run_sum[v] = run_sum[v] * scale + rowsum16(e0 + e1);
      acc0[v] *= scale; acc1[v] *= scale; acc2[v] *= scale; acc3[v] *= scale;
      int m = v + madd;
      myp[m * 32 + nrow]      = f2bf(e0);      // P tile C-layout -> LDS
      myp[m * 32 + 16 + nrow] = f2bf(e1);
    }
    asm volatile("s_wait_dscnt 0x0" ::: "memory");   // cross-lane LDS RAW
    v16bf pa = load_a_frag(myp, 32, 0, 0, lane);     // reload as A-fragment
    {
      v16bf vb = load_b_frag(Vbh, S_LEN, 0, s0, lane);
      acc0 = wmma_bf16(pa, vb, acc0);
      vb = load_b_frag(Vbh, S_LEN, 16, s0, lane);
      acc1 = wmma_bf16(pa, vb, acc1);
      vb = load_b_frag(Vbh, S_LEN, 32, s0, lane);
      acc2 = wmma_bf16(pa, vb, acc2);
      vb = load_b_frag(Vbh, S_LEN, 48, s0, lane);
      acc3 = wmma_bf16(pa, vb, acc3);
    }
    asm volatile("s_wait_dscnt 0x0" ::: "memory");   // LDS WAR before next store
  }

#pragma unroll
  for (int v = 0; v < 8; ++v) {
    float inv = 1.0f / run_sum[v];
    int tq = q0 + v + madd;
    size_t r = (size_t)tq * B_SZ + b;
    __bf16* crow = ctx + r * E_DIM + h * HD;
    crow[0 + nrow]  = f2bf(acc0[v] * inv);
    crow[16 + nrow] = f2bf(acc1[v] * inv);
    crow[32 + nrow] = f2bf(acc2[v] * inv);
    crow[48 + nrow] = f2bf(acc3[v] * inv);
  }
}

// ---- out-projection + residual (fp32) -------------------------------------
__global__ void __launch_bounds__(256)
outproj_kernel(const __bf16* __restrict__ A, const __bf16* __restrict__ W,
               const float* __restrict__ bias, const float* __restrict__ resid,
               float* __restrict__ out) {
  const int lane = threadIdx.x & 31;
  const int wid = blockIdx.x * 8 + (threadIdx.x >> 5);
  const int row0 = (wid >> 4) * 32;
  const int col0 = (wid & 15) * 64;
  v8f acc[8] = {};
  gemm_tile_32x64(A, E_DIM, row0, W, E_DIM, col0, E_DIM, lane, acc);
  const int nrow = lane & 15, madd = (lane >> 4) * 8;
#pragma unroll
  for (int t = 0; t < 4; ++t) {
    const int c = col0 + 16 * t + nrow;
    const float bc = bias[c];
#pragma unroll
    for (int g = 0; g < 2; ++g) {
#pragma unroll
      for (int v = 0; v < 8; ++v) {
        const size_t r = row0 + 16 * g + v + madd;
        out[r * E_DIM + c] = acc[g * 4 + t][v] + bc + resid[r * E_DIM + c];
      }
    }
  }
}

// ---- FFN1: relu(x @ w1^T + b1) -> bf16 ------------------------------------
__global__ void __launch_bounds__(256)
ffn1_kernel(const __bf16* __restrict__ A, const __bf16* __restrict__ W,
            const float* __restrict__ bias, __bf16* __restrict__ out) {
  const int lane = threadIdx.x & 31;
  const int wid = blockIdx.x * 8 + (threadIdx.x >> 5);
  const int row0 = (wid / 64) * 32;
  const int col0 = (wid % 64) * 64;
  v8f acc[8] = {};
  gemm_tile_32x64(A, E_DIM, row0, W, E_DIM, col0, E_DIM, lane, acc);
  const int nrow = lane & 15, madd = (lane >> 4) * 8;
#pragma unroll
  for (int t = 0; t < 4; ++t) {
    const int c = col0 + 16 * t + nrow;
    const float bc = bias[c];
#pragma unroll
    for (int g = 0; g < 2; ++g) {
#pragma unroll
      for (int v = 0; v < 8; ++v) {
        const size_t r = row0 + 16 * g + v + madd;
        out[r * F_DIM + c] = f2bf(fmaxf(acc[g * 4 + t][v] + bc, 0.0f));
      }
    }
  }
}

// ---- FFN2: h @ w2^T + b2 + residual(x) -> fp32 ----------------------------
__global__ void __launch_bounds__(256)
ffn2_kernel(const __bf16* __restrict__ A, const __bf16* __restrict__ W,
            const float* __restrict__ bias, const float* __restrict__ resid,
            float* __restrict__ out) {
  const int lane = threadIdx.x & 31;
  const int wid = blockIdx.x * 8 + (threadIdx.x >> 5);
  const int row0 = (wid >> 4) * 32;
  const int col0 = (wid & 15) * 64;
  v8f acc[8] = {};
  gemm_tile_32x64(A, F_DIM, row0, W, F_DIM, col0, F_DIM, lane, acc);
  const int nrow = lane & 15, madd = (lane >> 4) * 8;
#pragma unroll
  for (int t = 0; t < 4; ++t) {
    const int c = col0 + 16 * t + nrow;
    const float bc = bias[c];
#pragma unroll
    for (int g = 0; g < 2; ++g) {
#pragma unroll
      for (int v = 0; v < 8; ++v) {
        const size_t r = row0 + 16 * g + v + madd;
        out[r * E_DIM + c] = acc[g * 4 + t][v] + bc + resid[r * E_DIM + c];
      }
    }
  }
}

// ---- LayerNorm over E=1024, one block (256 thr) per row -------------------
__global__ void __launch_bounds__(256)
ln_kernel(const float* __restrict__ y, const float* __restrict__ g,
          const float* __restrict__ bta, float* __restrict__ xf,
          __bf16* __restrict__ xbf) {
  __shared__ float redS[8], redS2[8];
  const int row = blockIdx.x, tid = threadIdx.x;
  const float* yr = y + (size_t)row * E_DIM;
  float vals[4], s = 0.f, s2 = 0.f;
#pragma unroll
  for (int i = 0; i < 4; ++i) {
    float v = yr[tid + 256 * i];
    vals[i] = v; s += v; s2 += v * v;
  }
#pragma unroll
  for (int off = 16; off >= 1; off >>= 1) {
    s += __shfl_xor(s, off); s2 += __shfl_xor(s2, off);
  }
  if ((tid & 31) == 0) { redS[tid >> 5] = s; redS2[tid >> 5] = s2; }
  __syncthreads();
  s = 0.f; s2 = 0.f;
#pragma unroll
  for (int w = 0; w < 8; ++w) { s += redS[w]; s2 += redS2[w]; }
  float mu = s * (1.0f / E_DIM);
  float var = s2 * (1.0f / E_DIM) - mu * mu;
  float rstd = rsqrtf(var + 1e-5f);
#pragma unroll
  for (int i = 0; i < 4; ++i) {
    int c = tid + 256 * i;
    float o = (vals[i] - mu) * rstd * g[c] + bta[c];
    xf[(size_t)row * E_DIM + c] = o;
    if (xbf) xbf[(size_t)row * E_DIM + c] = f2bf(o);
  }
}

extern "C" void kernel_launch(void* const* d_in, const int* in_sizes, int n_in,
                              void* d_out, int out_size, void* d_ws, size_t ws_size,
                              hipStream_t stream) {
  (void)in_sizes; (void)n_in; (void)out_size; (void)ws_size;
  const float* src   = (const float*)d_in[0];
  const float* wqkv  = (const float*)d_in[1];
  const float* bqkv  = (const float*)d_in[2];
  const float* wo    = (const float*)d_in[3];
  const float* bo    = (const float*)d_in[4];
  const float* mem_k = (const float*)d_in[5];
  const float* mem_v = (const float*)d_in[6];
  const float* w1    = (const float*)d_in[7];
  const float* b1    = (const float*)d_in[8];
  const float* w2    = (const float*)d_in[9];
  const float* b2    = (const float*)d_in[10];
  const float* ln1g  = (const float*)d_in[11];
  const float* ln1b  = (const float*)d_in[12];
  const float* ln2g  = (const float*)d_in[13];
  const float* ln2b  = (const float*)d_in[14];

  char* ws = (char*)d_ws;
  size_t off = 0;
  auto take = [&](size_t bytes) -> char* {
    char* p = ws + off;
    off += (bytes + 255) & ~(size_t)255;
    return p;
  };
  __bf16* srcbf = (__bf16*)take((size_t)R_ROWS * E_DIM * 2);
  __bf16* wqb   = (__bf16*)take((size_t)3 * E_DIM * E_DIM * 2);
  __bf16* wob   = (__bf16*)take((size_t)E_DIM * E_DIM * 2);
  __bf16* w1b   = (__bf16*)take((size_t)F_DIM * E_DIM * 2);
  __bf16* w2b   = (__bf16*)take((size_t)E_DIM * F_DIM * 2);
  __bf16* Qbuf  = (__bf16*)take((size_t)B_SZ * H_NUM * T_SEQ * HD * 2);
  __bf16* Kbuf  = (__bf16*)take((size_t)B_SZ * H_NUM * S_LEN * HD * 2);
  __bf16* VTbuf = (__bf16*)take((size_t)B_SZ * H_NUM * HD * S_LEN * 2);
  __bf16* ctxb  = (__bf16*)take((size_t)R_ROWS * E_DIM * 2);
  float*  y1    = (float*)take((size_t)R_ROWS * E_DIM * 4);
  float*  xf    = (float*)take((size_t)R_ROWS * E_DIM * 4);
  __bf16* xbf   = (__bf16*)take((size_t)R_ROWS * E_DIM * 2);
  __bf16* hbuf  = (__bf16*)take((size_t)R_ROWS * F_DIM * 2);

  auto cvt = [&](const float* in, __bf16* out, int n) {
    f2bf_kernel<<<(n + 255) / 256, 256, 0, stream>>>(in, out, n);
  };
  cvt(src, srcbf, R_ROWS * E_DIM);
  cvt(wqkv, wqb, 3 * E_DIM * E_DIM);
  cvt(wo, wob, E_DIM * E_DIM);
  cvt(w1, w1b, F_DIM * E_DIM);
  cvt(w2, w2b, E_DIM * F_DIM);

  // 32-row tiles: rows/32 = 128 row-tiles for all GEMMs
  qkv_kernel<<<(128 * 48) / 8, 256, 0, stream>>>(srcbf, wqb, bqkv, Qbuf, Kbuf, VTbuf);
  memkv_kernel<<<128, 256, 0, stream>>>(mem_k, mem_v, Kbuf, VTbuf);
  attn_kernel<<<B_SZ * H_NUM * (T_SEQ / 64), 128, 0, stream>>>(Qbuf, Kbuf, VTbuf, ctxb);
  outproj_kernel<<<(128 * 16) / 8, 256, 0, stream>>>(ctxb, wob, bo, src, y1);
  ln_kernel<<<R_ROWS, 256, 0, stream>>>(y1, ln1g, ln1b, xf, xbf);
  ffn1_kernel<<<(128 * 64) / 8, 256, 0, stream>>>(xbf, w1b, b1, hbuf);
  ffn2_kernel<<<(128 * 16) / 8, 256, 0, stream>>>(hbuf, w2b, b2, xf, y1);   // y1 reused
  ln_kernel<<<R_ROWS, 256, 0, stream>>>(y1, ln2g, ln2b, (float*)d_out, (__bf16*)nullptr);
}
